// XDeepFM_18717467476536
// MI455X (gfx1250) — matrix operand order repaired and checked by reading, not compiled, not run on hardware
//
#include <hip/hip_runtime.h>
#include <cstdint>

typedef __attribute__((ext_vector_type(16))) __bf16 bf16x16;
typedef __attribute__((ext_vector_type(8)))  __bf16 bf16x8;
typedef __attribute__((ext_vector_type(8)))  float  f32x8;

namespace {
constexpr int kF0   = 39;
constexpr int kD    = 16;
constexpr int kL    = 128;
constexpr int kNC0  = 49;                 // layer-0 K chunks (K = i*40+j padded to 1568)
constexpr int kNC1  = 78;                 // layer-1 K chunks (K = i*64+j, 2496 exact)
constexpr int kNCT  = kNC0 + kNC1;        // 127
constexpr int kChunkElems = kL * 32;      // 4096 bf16 per chunk
constexpr int kChunkBytes = kChunkElems * 2;  // 8192 B
constexpr int kWaves = 8;
constexpr int kB     = 8192;
}

// ---------------- weight pre-transform: f32 [K,L] -> bf16 chunk-major [c][l][32]
__global__ void cin_prep_weights(const float* __restrict__ W0,
                                 const float* __restrict__ W1,
                                 __bf16* __restrict__ wp0,
                                 __bf16* __restrict__ wp1) {
  int e = blockIdx.x * blockDim.x + threadIdx.x;
  const int N0 = kNC0 * kChunkElems;
  const int N1 = kNC1 * kChunkElems;
  if (e < N0) {
    int ks = e & 31;
    int l  = (e >> 5) & 127;
    int c  = e >> 12;                 // / (32*128)
    int K  = c * 32 + ks;
    int i  = K / 40;
    int j  = K - i * 40;
    float v = 0.0f;
    if (i < kF0 && j < kF0) v = W0[(i * kF0 + j) * kL + l];
    wp0[e] = (__bf16)v;
  } else if (e < N0 + N1) {
    int e1 = e - N0;
    int ks = e1 & 31;
    int l  = (e1 >> 5) & 127;
    int c  = e1 >> 12;
    int K  = c * 32 + ks;
    int i  = K >> 6;
    int j  = K & 63;
    wp1[e1] = (__bf16)W1[(i * 64 + j) * kL + l];
  }
}

// ---------------- main CIN kernel: 8 waves/block, 1 batch element per wave
__launch_bounds__(256)
__global__ void cin_main(const float* __restrict__ x,
                         const __bf16* __restrict__ wp0,
                         const __bf16* __restrict__ wp1,
                         float* __restrict__ out) {
  __shared__ __align__(32) __bf16 wbuf[2][kChunkElems];   // 16 KB double-buffered weights
  __shared__ __align__(16) float  xT[kWaves][kD * 40];    // x[b]^T, row padded to 40
  __shared__ __align__(16) __bf16 HT[kWaves][kD * 64];    // relu(out0[:, :64])^T as bf16

  const int tid  = threadIdx.x;
  const int w    = tid >> 5;
  const int lane = tid & 31;
  const int n    = lane & 15;   // A-row d == lane%16; B/C column == lane%16
  const int h    = lane >> 4;   // lane half
  const int b    = blockIdx.x * kWaves + w;

  // stage x[b]^T into LDS (transpose [39][16] -> [16][40], zero pad col 39)
  const float* xb = x + (size_t)b * (kF0 * kD);
  for (int e = lane; e < kF0 * kD; e += 32) {
    int i = e >> 4;
    int d = e & 15;
    xT[w][d * 40 + i] = xb[e];
  }
  if (lane < 16) xT[w][lane * 40 + 39] = 0.0f;

  // async weight-chunk fetch: 8 KB / 256 threads = 2 x b128 per thread
  auto issue = [&](int k) {
    const char* src = (k < kNC0)
        ? ((const char*)wp0 + (size_t)k * kChunkBytes)
        : ((const char*)wp1 + (size_t)(k - kNC0) * kChunkBytes);
    unsigned ldsoff = (unsigned)(uintptr_t)(&wbuf[k & 1][0]) + (unsigned)(tid * 32);
    unsigned voff   = (unsigned)(tid * 32);
    asm volatile(
        "global_load_async_to_lds_b128 %0, %1, %2\n\t"
        "global_load_async_to_lds_b128 %0, %1, %2 offset:16"
        :: "v"(ldsoff), "v"(voff), "s"(src)
        : "memory");
  };

  const f32x8 vzero = {};
  f32x8 acc[8] = {};

  issue(0);
  issue(1);

  for (int cc = 0; cc < kNCT; ++cc) {
    // wait for chunk cc to land (2 async instrs per chunk per wave, in-order)
    if (cc + 1 < kNCT) {
      asm volatile("s_wait_asynccnt 0x2" ::: "memory");
    } else {
      asm volatile("s_wait_asynccnt 0x0" ::: "memory");
    }
    __syncthreads();
    const int pb = cc & 1;

    // ---- build A fragment (16 bf16 per lane): K-set {base+0..15}
    bf16x16 a;
    if (cc < kNC0) {
      // layer 0: A[d,K] = x[i,d]*x[j,d], K = i*40 + j (aligned 8-runs: i const, j contiguous)
      #pragma unroll
      for (int s2 = 0; s2 < 2; ++s2) {
        int kb = cc * 32 + h * 16 + s2 * 8;
        int i  = kb / 40;
        int j0 = kb - i * 40;
        float  xi = xT[w][n * 40 + i];
        float4 v0 = *(const float4*)&xT[w][n * 40 + j0];
        float4 v1 = *(const float4*)&xT[w][n * 40 + j0 + 4];
        a[s2 * 8 + 0] = (__bf16)(xi * v0.x);
        a[s2 * 8 + 1] = (__bf16)(xi * v0.y);
        a[s2 * 8 + 2] = (__bf16)(xi * v0.z);
        a[s2 * 8 + 3] = (__bf16)(xi * v0.w);
        a[s2 * 8 + 4] = (__bf16)(xi * v1.x);
        a[s2 * 8 + 5] = (__bf16)(xi * v1.y);
        a[s2 * 8 + 6] = (__bf16)(xi * v1.z);
        a[s2 * 8 + 7] = (__bf16)(xi * v1.w);
      }
    } else {
      // layer 1: A[d,K] = x[i,d]*H[j,d], K = i*64 + j
      int c1 = cc - kNC0;
      #pragma unroll
      for (int s2 = 0; s2 < 2; ++s2) {
        int kb = c1 * 32 + h * 16 + s2 * 8;
        int i  = kb >> 6;
        int j0 = kb & 63;
        float  xi = xT[w][n * 40 + i];
        bf16x8 hv = *(const bf16x8*)&HT[w][n * 64 + j0];
        #pragma unroll
        for (int t = 0; t < 8; ++t)
          a[s2 * 8 + t] = (__bf16)(xi * (float)hv[t]);
      }
    }

    // ---- 8 output l-tiles: stage ALL B-fragments in registers first, then run
    // the WMMA chain. The sched_barrier keeps the 16 ds_load_b128 grouped ahead
    // of the first v_wmma so dscnt waits pipeline (0xe,0xc,...) instead of 8x
    // full drains; LDS latency overlaps matrix-core execution.
    bf16x16 bv[8];
    #pragma unroll
    for (int lt = 0; lt < 8; ++lt)
      bv[lt] = *(const bf16x16*)&wbuf[pb][(lt * 16 + n) * 32 + h * 16];
    __builtin_amdgcn_sched_barrier(0);
    #pragma unroll
    for (int lt = 0; lt < 8; ++lt)
      acc[lt] = __builtin_amdgcn_wmma_f32_16x16x32_bf16(
          false, a, false, bv[lt], (short)0, acc[lt], false, false);

    // ---- layer-0 epilogue: H^T -> LDS (tiles 0..3), direct0 reduce (tiles 4..7)
    if (cc == kNC0 - 1) {
      #pragma unroll
      for (int lt = 0; lt < 4; ++lt) {
        #pragma unroll
        for (int r = 0; r < 8; ++r)
          HT[w][(r + 8 * h) * 64 + lt * 16 + n] = (__bf16)fmaxf(acc[lt][r], 0.0f);
      }
      #pragma unroll
      for (int lt = 4; lt < 8; ++lt) {
        float s = 0.0f;
        #pragma unroll
        for (int r = 0; r < 8; ++r) s += fmaxf(acc[lt][r], 0.0f);
        s += __shfl_xor(s, 16, 32);
        if (h == 0) out[(size_t)b * 192 + (lt - 4) * 16 + n] = s;
      }
      #pragma unroll
      for (int t = 0; t < 8; ++t) acc[t] = vzero;
    }

    __syncthreads();                      // everyone done reading buf pb
    if (cc + 2 < kNCT) issue(cc + 2);     // refill it
  }

  // ---- layer-1 epilogue: relu + sum over d
  #pragma unroll
  for (int lt = 0; lt < 8; ++lt) {
    float s = 0.0f;
    #pragma unroll
    for (int r = 0; r < 8; ++r) s += fmaxf(acc[lt][r], 0.0f);
    s += __shfl_xor(s, 16, 32);
    if (h == 0) out[(size_t)b * 192 + 64 + lt * 16 + n] = s;
  }
}

extern "C" void kernel_launch(void* const* d_in, const int* in_sizes, int n_in,
                              void* d_out, int out_size, void* d_ws, size_t ws_size,
                              hipStream_t stream) {
  (void)in_sizes; (void)n_in; (void)out_size; (void)ws_size;
  const float* hx = (const float*)d_in[0];
  const float* W0 = (const float*)d_in[1];
  const float* W1 = (const float*)d_in[2];
  float* out = (float*)d_out;

  __bf16* wp0 = (__bf16*)d_ws;
  __bf16* wp1 = wp0 + (size_t)kNC0 * kChunkElems;   // ~1.04 MB total in d_ws

  const int total = kNCT * kChunkElems;
  cin_prep_weights<<<(total + 255) / 256, 256, 0, stream>>>(W0, W1, wp0, wp1);
  cin_main<<<kB / kWaves, 256, 0, stream>>>(hx, wp0, wp1, out);
}
